// Attention_18339510354363
// MI455X (gfx1250) — matrix operand order repaired and checked by reading, not compiled, not run on hardware
//
#include <hip/hip_runtime.h>

// ---------------------------------------------------------------------------
// Problem constants (from reference): B=8, N=4096, C=768, H=8, Ch=96
// ---------------------------------------------------------------------------
#define BB   8
#define NTOK 4096
#define CDIM 768
#define HN   8
#define CHD  96
#define QKVW (3 * CDIM)   // 2304

typedef __attribute__((ext_vector_type(16))) __bf16 v16bf;
typedef __attribute__((ext_vector_type(8)))  float  v8f;

union Frag {
    uint32_t u[8];
    uint4    q[2];
    v16bf    v;
};

__device__ __forceinline__ unsigned short f2bf(float f) {
    union { float f; uint32_t u; } x; x.f = f;
    uint32_t u = x.u;
    u += 0x7FFFu + ((u >> 16) & 1u);   // round-to-nearest-even
    return (unsigned short)(u >> 16);
}
__device__ __forceinline__ uint32_t f2bf_pk(float lo, float hi) {
    return (uint32_t)f2bf(lo) | ((uint32_t)f2bf(hi) << 16);
}
__device__ __forceinline__ float bf2f(unsigned short h) {
    union { uint32_t u; float f; } x; x.u = ((uint32_t)h) << 16;
    return x.f;
}

// A-fragment pair order: reg j (half h) reads k-pair
//   pp = h*8 + j   (contiguous, same as B)
// so aslot is the identity.
__device__ __forceinline__ constexpr int aslot(int pp) {
    return pp;
}

__device__ __forceinline__ v8f wmma_bf16(const Frag& a, const Frag& b, v8f c) {
    return __builtin_amdgcn_wmma_f32_16x16x32_bf16(
        false, a.v, false, b.v, (short)0, c, false, false);
}

// ---------------------------------------------------------------------------
// fp32 -> bf16 conversion kernel (vectorized x4)
// ---------------------------------------------------------------------------
__global__ void cvt_f32_to_bf16_x4(const float4* __restrict__ in,
                                   uint2* __restrict__ out, int n4) {
    int i = blockIdx.x * blockDim.x + threadIdx.x;
    if (i < n4) {
        float4 f = in[i];
        uint2 o;
        o.x = f2bf_pk(f.x, f.y);
        o.y = f2bf_pk(f.z, f.w);
        out[i] = o;
    }
}

// ---------------------------------------------------------------------------
// Generic bf16 GEMM: D[M,N] = A[M,K] * B[K,N] (+ bias[N]).
// 256 threads (8 waves). Block tile 128x128, K-step 32, double-buffered LDS
// (one barrier per K-step; next slab's global loads + LDS stores overlap the
// current slab's ds_load/v_wmma stream).
// Wave grid 4x2; each wave computes a 32x64 region = 2x4 WMMA tiles.
// LDS layouts are fragment-swizzled:
//   As[buf][m][slot]   u32 k-pairs, aslot-permuted -> A frag = contiguous LDS
//   Bs[buf][n][kpair]  u32 k-pairs, natural order  -> B frag = contiguous LDS
// Requires M%128==0, N%128==0, K%32==0 (true for all calls here).
// ---------------------------------------------------------------------------
template <typename OutT, bool HAS_BIAS>
__global__ __launch_bounds__(256)
void gemm_bf16_128x128(const unsigned short* __restrict__ A,
                       const unsigned short* __restrict__ B,
                       OutT* __restrict__ D,
                       const float* __restrict__ bias,
                       int M, int N, int K) {
    __shared__ uint32_t As[2][128][16];   // 16 KB
    __shared__ uint32_t Bs[2][128][16];   // 16 KB

    const int tid  = threadIdx.x;
    const int w    = tid >> 5;
    const int lane = tid & 31;
    const int half = lane >> 4;
    const int l15  = lane & 15;
    const int wr   = w >> 1;          // 0..3 -> m offset wr*32
    const int wc   = w & 1;           // 0..1 -> n offset wc*64
    const int bm   = blockIdx.y * 128;
    const int bn   = blockIdx.x * 128;

    v8f acc[2][4];
#pragma unroll
    for (int i = 0; i < 2; ++i)
#pragma unroll
        for (int j = 0; j < 4; ++j) { v8f z = {}; acc[i][j] = z; }

    // A staging: thread -> (row am, 8 k-pairs starting at app)
    const int am  = tid >> 1;            // 0..127
    const int app = (tid & 1) * 8;       // pair base 0 or 8
    // B staging: thread -> (k-pair bkp, 8 n starting at bn0)
    const int bkp = tid >> 4;            // 0..15
    const int bn0 = (tid & 15) * 8;      // 0..120

    union { uint4 v[2]; unsigned short s[16]; } ta;
    union { uint4 v; unsigned short s[8]; } tb0, tb1;

    // ---- prologue: load + stage step 0 -------------------------------------
    {
        const uint4* ap = (const uint4*)(A + (size_t)(bm + am) * K + app * 2);
        ta.v[0] = ap[0]; ta.v[1] = ap[1];
        const size_t brow = (size_t)(bkp * 2) * N + bn + bn0;
        tb0.v = *(const uint4*)(B + brow);
        tb1.v = *(const uint4*)(B + brow + N);
#pragma unroll
        for (int i = 0; i < 8; ++i) {
            As[0][am][aslot(app + i)] =
                (uint32_t)ta.s[2 * i] | ((uint32_t)ta.s[2 * i + 1] << 16);
            Bs[0][bn0 + i][bkp] =
                (uint32_t)tb0.s[i] | ((uint32_t)tb1.s[i] << 16);
        }
    }
    __syncthreads();

    const int nsteps = K >> 5;
    for (int step = 0; step < nsteps; ++step) {
        const int cur  = step & 1;
        const int nxt  = cur ^ 1;
        const bool more = (step + 1) < nsteps;

        if (more) {
            const int kb = (step + 1) * 32;
            const uint4* ap =
                (const uint4*)(A + (size_t)(bm + am) * K + kb + app * 2);
            ta.v[0] = ap[0]; ta.v[1] = ap[1];
            const size_t brow = (size_t)(kb + bkp * 2) * N + bn + bn0;
            tb0.v = *(const uint4*)(B + brow);
            tb1.v = *(const uint4*)(B + brow + N);
#pragma unroll
            for (int i = 0; i < 8; ++i) {
                As[nxt][am][aslot(app + i)] =
                    (uint32_t)ta.s[2 * i] | ((uint32_t)ta.s[2 * i + 1] << 16);
                Bs[nxt][bn0 + i][bkp] =
                    (uint32_t)tb0.s[i] | ((uint32_t)tb1.s[i] << 16);
            }
        }

        Frag af[2];
#pragma unroll
        for (int mi = 0; mi < 2; ++mi) {
            const int m = wr * 32 + mi * 16 + l15;
            af[mi].q[0] = *(const uint4*)&As[cur][m][half * 8];
            af[mi].q[1] = *(const uint4*)&As[cur][m][half * 8 + 4];
        }
#pragma unroll
        for (int ni = 0; ni < 4; ++ni) {
            Frag bf;
            const int n = wc * 64 + ni * 16 + l15;
            bf.q[0] = *(const uint4*)&Bs[cur][n][half * 8];
            bf.q[1] = *(const uint4*)&Bs[cur][n][half * 8 + 4];
#pragma unroll
            for (int mi = 0; mi < 2; ++mi)
                acc[mi][ni] = wmma_bf16(af[mi], bf, acc[mi][ni]);
        }
        __syncthreads();
    }

    // epilogue: C/D layout -> m = half*8 + r, n = l15 within each 16x16 tile
#pragma unroll
    for (int mi = 0; mi < 2; ++mi) {
#pragma unroll
        for (int ni = 0; ni < 4; ++ni) {
            const int mg = bm + wr * 32 + mi * 16 + half * 8;
            const int ng = bn + wc * 64 + ni * 16 + l15;
            float bv = 0.0f;
            if constexpr (HAS_BIAS) bv = bias[ng];
#pragma unroll
            for (int r = 0; r < 8; ++r) {
                const float v = acc[mi][ni][r] + bv;
                OutT* p = D + (size_t)(mg + r) * N + ng;
                if constexpr (sizeof(OutT) == 2) *p = (OutT)f2bf(v);
                else                             *p = (OutT)v;
            }
        }
    }
}

// ---------------------------------------------------------------------------
// Fused channel attention, one workgroup per (b,h). 192 threads = 6 waves.
// qkv layout: row (b,n) of 2304 bf16; col = s*768 + h*96 + c  (s=0:q,1:k,2:v)
// Phase A: 1/max(||q_c||,eps), 1/max(||k_c||,eps) over n (fp32)
// Phase B: S[96][96] = q k^T over n via WMMA (raw), post-scaled by rq*rk*tau.
//          Double-buffered LDS staging; each thread stages a token-pair x 8
//          channels as packed u32 (ds_store_b32) in fragment-slot layout.
// Phase C: fp32 softmax over d, P stored bf16 (A-frag swizzled)
// Phase D: out[c][n] = P V, streamed over 256 n-tiles, bf16 out to (B,N,C)
// ---------------------------------------------------------------------------
__global__ __launch_bounds__(192)
void channel_attn_kernel(const unsigned short* __restrict__ qkv,
                         const float* __restrict__ temperature,
                         unsigned short* __restrict__ out2) {
    const int b = blockIdx.x >> 3;
    const int h = blockIdx.x & 7;

    __shared__ union {
        struct {
            uint32_t Qs[2][96][16];   // [buf][c][slot]  A-frag swizzled, 12 KB
            uint32_t Ks[2][96][16];   // [buf][d][npair] B-frag layout,  12 KB
        } g;
        float Sf[96][96];             // 36 KB
    } sh;
    __shared__ uint32_t Pp[96][48];   // 18 KB, per-32k-group A-frag swizzled
    __shared__ float red[192];
    __shared__ float rqs[96], rks[96];

    const int t    = threadIdx.x;
    const int w    = t >> 5;
    const int lane = t & 31;
    const int half = lane >> 4;
    const int l15  = lane & 15;
    const size_t rowbase = (size_t)(b * NTOK) * QKVW;

    // ---- Phase A: inverse norms --------------------------------------------
    {
        const int c = t % 96, p = t / 96;       // p in {0,1}
        float sq = 0.f, sk = 0.f;
        for (int n = p * 2048; n < p * 2048 + 2048; ++n) {
            const size_t base = rowbase + (size_t)n * QKVW + h * CHD + c;
            const float qv = bf2f(qkv[base]);
            const float kv = bf2f(qkv[base + CDIM]);
            sq += qv * qv;
            sk += kv * kv;
        }
        red[t] = sq; __syncthreads();
        if (t < 96) rqs[t] = 1.0f / fmaxf(sqrtf(red[t] + red[t + 96]), 1e-12f);
        __syncthreads();
        red[t] = sk; __syncthreads();
        if (t < 96) rks[t] = 1.0f / fmaxf(sqrtf(red[t] + red[t + 96]), 1e-12f);
        __syncthreads();
    }

    // ---- Phase B: Gram matrix S = q k^T over n (K = 4096, step 32) ---------
    v8f acc[6];
#pragma unroll
    for (int i = 0; i < 6; ++i) { v8f z = {}; acc[i] = z; }

    // staging: thread -> token-pair tp (0..15) x 8 channels at cg
    const int tp = t & 15;            // token pair within 32-token slab
    const int cg = (t >> 4) * 8;      // channel group base: 0,8,...,88
    const int qslot = aslot(tp);      // Q (A-operand) slot for this pair
    union { uint4 v; unsigned short s[8]; } q0, q1, k0, k1;

    // prologue: load + stage slab 0 into buffer 0
    {
        const size_t gb = rowbase + (size_t)(2 * tp) * QKVW + h * CHD + cg;
        q0.v = *(const uint4*)(qkv + gb);
        q1.v = *(const uint4*)(qkv + gb + QKVW);
        k0.v = *(const uint4*)(qkv + gb + CDIM);
        k1.v = *(const uint4*)(qkv + gb + CDIM + QKVW);
#pragma unroll
        for (int i = 0; i < 8; ++i) {
            sh.g.Qs[0][cg + i][qslot] =
                (uint32_t)q0.s[i] | ((uint32_t)q1.s[i] << 16);
            sh.g.Ks[0][cg + i][tp] =
                (uint32_t)k0.s[i] | ((uint32_t)k1.s[i] << 16);
        }
    }
    __syncthreads();

    for (int step = 0; step < 128; ++step) {
        const int cur  = step & 1;
        const int nxt  = cur ^ 1;
        const bool more = (step + 1) < 128;

        if (more) {
            const int nb = (step + 1) * 32;
            const size_t gb =
                rowbase + (size_t)(nb + 2 * tp) * QKVW + h * CHD + cg;
            q0.v = *(const uint4*)(qkv + gb);
            q1.v = *(const uint4*)(qkv + gb + QKVW);
            k0.v = *(const uint4*)(qkv + gb + CDIM);
            k1.v = *(const uint4*)(qkv + gb + CDIM + QKVW);
#pragma unroll
            for (int i = 0; i < 8; ++i) {
                sh.g.Qs[nxt][cg + i][qslot] =
                    (uint32_t)q0.s[i] | ((uint32_t)q1.s[i] << 16);
                sh.g.Ks[nxt][cg + i][tp] =
                    (uint32_t)k0.s[i] | ((uint32_t)k1.s[i] << 16);
            }
        }

        Frag af;
        const int mrow = 16 * w + l15;
        af.q[0] = *(const uint4*)&sh.g.Qs[cur][mrow][half * 8];
        af.q[1] = *(const uint4*)&sh.g.Qs[cur][mrow][half * 8 + 4];
#pragma unroll
        for (int dt = 0; dt < 6; ++dt) {
            Frag bf;
            const int d = 16 * dt + l15;
            bf.q[0] = *(const uint4*)&sh.g.Ks[cur][d][half * 8];
            bf.q[1] = *(const uint4*)&sh.g.Ks[cur][d][half * 8 + 4];
            acc[dt] = wmma_bf16(af, bf, acc[dt]);
        }
        __syncthreads();
    }

    // post-scale: normalization + temperature, write fp32 S (aliases staging)
    {
        const float tau = temperature[h];
#pragma unroll
        for (int dt = 0; dt < 6; ++dt) {
#pragma unroll
            for (int r = 0; r < 8; ++r) {
                const int m = 16 * w + half * 8 + r;
                const int d = 16 * dt + l15;
                sh.Sf[m][d] = acc[dt][r] * rqs[m] * rks[d] * tau;
            }
        }
    }
    __syncthreads();

    // ---- Phase C: softmax over d, P -> bf16 (A-frag swizzled per 32-group) --
    if (t < 96) {
        float mx = -3.402823466e+38f;
        for (int d = 0; d < 96; ++d) mx = fmaxf(mx, sh.Sf[t][d]);
        float s = 0.f;
        for (int d = 0; d < 96; ++d) s += __expf(sh.Sf[t][d] - mx);
        const float inv = 1.0f / s;
        unsigned short* Ph = (unsigned short*)&Pp[0][0];
        for (int d = 0; d < 96; ++d) {
            const int f  = d >> 5;                 // 32-wide k group
            const int pl = (d >> 1) & 15;          // pair within group
            Ph[t * 96 + (f * 16 + aslot(pl)) * 2 + (d & 1)] =
                f2bf(__expf(sh.Sf[t][d] - mx) * inv);
        }
    }
    __syncthreads();

    // ---- Phase D: out[c][n] = P V ; stream 256 n-tiles of 16 ----------------
    Frag pf[3];
#pragma unroll
    for (int f = 0; f < 3; ++f) {
        pf[f].q[0] = *(const uint4*)&Pp[16 * w + l15][f * 16 + half * 8];
        pf[f].q[1] = *(const uint4*)&Pp[16 * w + l15][f * 16 + half * 8 + 4];
    }

    const size_t vbase = rowbase + 2 * CDIM + h * CHD;           // s=2 (v)
    const size_t obase = (size_t)(b * NTOK) * CDIM + h * CHD;

    for (int nt = 0; nt < 256; ++nt) {
        const int ng = nt * 16 + l15;
        v8f o = {};
#pragma unroll
        for (int f = 0; f < 3; ++f) {
            Frag bf;
            const int dbase = f * 32 + half * 16;
            const uint4* vp =
                (const uint4*)(qkv + vbase + (size_t)ng * QKVW + dbase);
            bf.q[0] = vp[0];
            bf.q[1] = vp[1];
            o = wmma_bf16(pf[f], bf, o);
        }
        union { uint4 v; unsigned short s[8]; } ou;
#pragma unroll
        for (int r = 0; r < 8; ++r) ou.s[r] = f2bf(o[r]);
        *(uint4*)(out2 + obase + (size_t)ng * CDIM + 16 * w + half * 8) = ou.v;
    }
}

// ---------------------------------------------------------------------------
// Launch: cvt(x,Wqkv,Wproj) -> GEMM(qkv) -> fused attention -> GEMM(proj+bias)
// ---------------------------------------------------------------------------
extern "C" void kernel_launch(void* const* d_in, const int* in_sizes, int n_in,
                              void* d_out, int out_size, void* d_ws, size_t ws_size,
                              hipStream_t stream) {
    const float* x     = (const float*)d_in[0];
    const float* Wqkv  = (const float*)d_in[1];
    const float* temp  = (const float*)d_in[2];
    const float* Wproj = (const float*)d_in[3];
    const float* bproj = (const float*)d_in[4];
    float* out = (float*)d_out;

    const size_t M = (size_t)BB * NTOK;      // 32768
    unsigned short* xb     = (unsigned short*)d_ws;
    unsigned short* wqkvb  = xb    + M * CDIM;
    unsigned short* wprojb = wqkvb + (size_t)CDIM * QKVW;
    unsigned short* qkvb   = wprojb + (size_t)CDIM * CDIM;
    unsigned short* out2b  = qkvb  + M * QKVW;

    const int nx4 = (int)(M * CDIM / 4);
    const int nq4 = CDIM * QKVW / 4;
    const int np4 = CDIM * CDIM / 4;
    cvt_f32_to_bf16_x4<<<(nx4 + 255) / 256, 256, 0, stream>>>(
        (const float4*)x, (uint2*)xb, nx4);
    cvt_f32_to_bf16_x4<<<(nq4 + 255) / 256, 256, 0, stream>>>(
        (const float4*)Wqkv, (uint2*)wqkvb, nq4);
    cvt_f32_to_bf16_x4<<<(np4 + 255) / 256, 256, 0, stream>>>(
        (const float4*)Wproj, (uint2*)wprojb, np4);

    // qkv = x @ Wqkv : M=32768, N=2304, K=768
    dim3 g1(QKVW / 128, (int)(M / 128));
    gemm_bf16_128x128<unsigned short, false>
        <<<g1, 256, 0, stream>>>(xb, wqkvb, qkvb, nullptr, (int)M, QKVW, CDIM);

    // fused channel attention: 64 workgroups (b,h), 6 waves each
    channel_attn_kernel<<<BB * HN, 192, 0, stream>>>(qkvb, temp, out2b);

    // out = out2 @ Wproj + bproj : M=32768, N=768, K=768 (fp32 out)
    dim3 g2(CDIM / 128, (int)(M / 128));
    gemm_bf16_128x128<float, true>
        <<<g2, 256, 0, stream>>>(out2b, wprojb, out, bproj, (int)M, CDIM, CDIM);
}
